// AlternatingForecastModel_68839735820745
// MI455X (gfx1250) — matrix operand order repaired and checked by reading, not compiled, not run on hardware
//
#include <hip/hip_runtime.h>

typedef __attribute__((ext_vector_type(16))) __bf16 v16bf;
typedef __attribute__((ext_vector_type(8)))  float  v8f;

#define BB 256
#define TT 4096
#define FF 16
#define HH 256
#define WW 10
#define NG 1024          // 4*H gate columns
#define E_F 2.718281828459045f

// workspace byte offsets
#define BHH_OFF   0u        // bf16 [64 nt][8 kt][32 lane][16 e]  = 512 KB
#define BIH_OFF   524288u   // bf16 [64 nt][32 lane][16 e]        = 64 KB
#define ANOM_OFF  589824u   // int  [4096]
#define MODE_OFF  606208u   // int  [4096]
#define HEAD_OFF  622592u   // int  [4096]
#define WNORM_OFF 638976u   // f32  [4096]
#define BIAS_OFF  655360u   // f32  [1024]  (b_ih + b_hh)

// ---- B-fragment layout for v_wmma_f32_16x16x32_bf16 (B = [K=32][N=16]):
// lanes 0-15: N=lane,    K = e       (e = 0..15)
// lanes16-31: N=lane-16, K = 16 + e
__global__ void pack_whh(const float* __restrict__ Whh, __bf16* __restrict__ Bp) {
  int nt = blockIdx.x, kt = blockIdx.y, lane = threadIdx.x;
  int n  = nt * 16 + (lane & 15);
  int kb = kt * 32 + ((lane >> 4) << 4);
  __bf16* dst = Bp + ((((nt * 8 + kt) * 32) + lane) << 4);
#pragma unroll
  for (int e = 0; e < 16; ++e) dst[e] = (__bf16)Whh[n * HH + kb + e];
}

// W_ih[:, 0:16] only (the appended zero column never contributes); K>=16 zero-padded.
__global__ void pack_wih(const float* __restrict__ Wih, __bf16* __restrict__ Bp) {
  int nt = blockIdx.x, lane = threadIdx.x;
  int n  = nt * 16 + (lane & 15);
  int kb = (lane >> 4) << 4;
  __bf16* dst = Bp + ((nt * 32 + lane) << 4);
#pragma unroll
  for (int e = 0; e < 16; ++e) {
    int k = kb + e;
    dst[e] = (k < FF) ? (__bf16)Wih[n * (FF + 1) + k] : (__bf16)0.0f;
  }
}

__global__ void bias_sum(const float* __restrict__ bih, const float* __restrict__ bhh,
                         float* __restrict__ bias) {
  int i = blockIdx.x * 256 + threadIdx.x;
  if (i < NG) bias[i] = bih[i] + bhh[i];
}

// anom[t] = any_b (x[b,t,F-1] != 0)  -- depends only on inputs, so precomputable
__global__ void anom_scan(const float* __restrict__ x, int* __restrict__ anom) {
  __shared__ int flag;
  if (threadIdx.x == 0) flag = 0;
  __syncthreads();
  int t = blockIdx.x;
  float v = x[((size_t)threadIdx.x * TT + t) * FF + (FF - 1)];
  if (v != 0.0f) atomicOr(&flag, 1);
  __syncthreads();
  if (threadIdx.x == 0) anom[t] = flag;
}

// Sequential per-step schedule: mode 0=anomalous, 1=warmup append(pos=head, wnorm=1/sum e^i),
// 2=full-buffer roll+append (O(1) incremental update of weighted sum S).
__global__ void schedule_k(const int* __restrict__ anom, int* __restrict__ mode,
                           int* __restrict__ headArr, float* __restrict__ wnorm) {
  int count = 1, head = 0;
  for (int t = 0; t < TT; ++t) {
    if (anom[t])            { mode[t] = 0; headArr[t] = 0;     wnorm[t] = 0.f; }
    else if (count == WW)   { mode[t] = 2; headArr[t] = head;  wnorm[t] = 0.f; head = (head + 1) % WW; }
    else {
      mode[t] = 1; headArr[t] = count;
      float s = 0.f, ei = 1.f;
      for (int i = 0; i <= count; ++i) { s += ei; ei *= E_F; }
      wnorm[t] = 1.f / s; ++count;
    }
  }
}

__device__ __forceinline__ float sigmoidf_(float v) { return 1.f / (1.f + __expf(-v)); }

// A-fragment index map (16-bit A matrix 16x32): element e of lane l holds
// K = e + (l>=16 ? 16 : 0); M = l&15.
__global__ __launch_bounds__(128) void lstm_main(
    const float* __restrict__ x, const __bf16* __restrict__ Bhh, const __bf16* __restrict__ Bih,
    const int* __restrict__ mode, const int* __restrict__ headArr, const float* __restrict__ wnorm,
    const float* __restrict__ bias, const float* __restrict__ Wout, const float* __restrict__ bout,
    float* __restrict__ out, float* __restrict__ hOut, float* __restrict__ cOut) {
  extern __shared__ unsigned char smem[];
  float*  ring   = (float*)smem;            // [10][16][257]  good-memory ring buffer
  float*  gates  = ring  + 10 * 16 * 257;   // [4][16][257]   i,f,g,o staging
  float*  cbuf   = gates + 4 * 16 * 257;    // [16][257]      cell state
  float*  Sbuf   = cbuf  + 16 * 257;        // [16][257]      softmax-weighted buffer sum
  float*  xs     = Sbuf  + 16 * 257;        // [16][16]       x_t tile
  float*  woutS  = xs    + 256;             // [256]
  float*  biasS  = woutS + 256;             // [1024]
  float*  outacc = biasS + 1024;            // [16]
  __bf16* hA     = (__bf16*)(outacc + 16);  // [8 kt][32 lane][16 e]  h in A-fragment layout

  const int tid  = threadIdx.x;
  const int wave = tid >> 5;                // wave g owns gate g (i/f/g/o), 256 columns
  const int lane = tid & 31;
  const int wgb  = blockIdx.x * 16;         // batch tile base
  const int bloc = tid >> 3;                // elementwise: batch row 0..15
  const int jbase = tid & 7;                // elementwise: 8 threads per row, 32 j's each

  for (int i = tid; i < 16 * 257 * 16; i += 128) ring[i] = 0.f;  // zero ring+gates+c+S
  for (int i = tid; i < 2048; i += 128) ((unsigned int*)hA)[i] = 0u;
  for (int i = tid; i < 256;  i += 128) woutS[i] = Wout[i];
  for (int i = tid; i < 1024; i += 128) biasS[i] = bias[i];
  __syncthreads();

  const float b0    = bout[0];
  const float inv_e = 1.f / E_F;
  float S10 = 0.f; { float ei = 1.f; for (int i = 0; i < WW; ++i) { S10 += ei; ei *= E_F; } }
  const float w0c = 1.f / S10;
  float w9c; { float ei = 1.f; for (int i = 0; i < 9; ++i) ei *= E_F; w9c = ei / S10; }

  // ---- Explicit VGPR-resident weights (fits the 1024-VGPR file; rest is streamed):
  //   bihr: 16 frags (128 VGPRs), bhhr: kt=0..3, 64 frags (512 VGPRs), acc: 128 VGPRs.
  v16bf bihr[16];
#pragma unroll
  for (int nt = 0; nt < 16; ++nt)
    bihr[nt] = *(const v16bf*)(Bih + (((wave * 16 + nt) * 32 + lane) << 4));
  v16bf bhhr[64];
#pragma unroll
  for (int kt = 0; kt < 4; ++kt)
#pragma unroll
    for (int nt = 0; nt < 16; ++nt)
      bhhr[kt * 16 + nt] =
          *(const v16bf*)(Bhh + ((((wave * 16 + nt) * 8 + kt) * 32 + lane) << 4));

  for (int t = 0; t < TT; ++t) {
    // ---- P1: stage x_t tile, prefetch x_{t+1}, clear output accumulators
    if (tid < 64) {
      int m = tid >> 2, f4 = (tid & 3) << 2;
      const float* xp = x + ((size_t)(wgb + m) * TT + t) * FF + f4;
      const float4 v4 = *(const float4*)xp;
      xs[m * 16 + f4 + 0] = v4.x; xs[m * 16 + f4 + 1] = v4.y;
      xs[m * 16 + f4 + 2] = v4.z; xs[m * 16 + f4 + 3] = v4.w;
      if (t + 1 < TT) __builtin_prefetch(xp + FF, 0, 1);   // global_prefetch_b8
    }
    if (tid < 16) outacc[tid] = 0.f;
    __syncthreads();

    // ---- P2: gates = x_t @ W_ih[:, :16].T  +  h @ W_hh.T   (bf16 WMMA, f32 acc)
    v8f acc[16];
    v16bf ax;
    {
      int m = lane & 15;
#pragma unroll
      for (int e = 0; e < 16; ++e) {
        int vv = e >> 1;
        int K  = ((lane >> 4) << 4) + e;
        ax[e]  = (K < FF) ? (__bf16)xs[m * 16 + K] : (__bf16)0.0f;
      }
    }
#pragma unroll
    for (int nt = 0; nt < 16; ++nt) {
      v8f z = {};
      acc[nt] = __builtin_amdgcn_wmma_f32_16x16x32_bf16(false, ax, false, bihr[nt], (short)0, z, false, false);
    }
    // resident half: kt = 0..3
#pragma unroll
    for (int kt = 0; kt < 4; ++kt) {
      v16bf ah = *(const v16bf*)(hA + ((kt * 32 + lane) << 4));
#pragma unroll
      for (int nt = 0; nt < 16; ++nt)
        acc[nt] = __builtin_amdgcn_wmma_f32_16x16x32_bf16(false, ah, false, bhhr[kt * 16 + nt], (short)0, acc[nt], false, false);
    }
    // streamed half: kt = 4..7.  Launder the base pointer through asm so these
    // loop-invariant loads CANNOT be hoisted (that hoist caused scratch spills).
    {
      const __bf16* bs = Bhh;
      asm volatile("" : "+s"(bs));
#pragma unroll
      for (int kt = 4; kt < 8; ++kt) {
        v16bf ah = *(const v16bf*)(hA + ((kt * 32 + lane) << 4));
#pragma unroll
        for (int nt = 0; nt < 16; ++nt) {
          v16bf bfr = *(const v16bf*)(bs + ((((wave * 16 + nt) * 8 + kt) * 32 + lane) << 4));
          acc[nt] = __builtin_amdgcn_wmma_f32_16x16x32_bf16(false, ah, false, bfr, (short)0, acc[nt], false, false);
        }
      }
    }
    {  // scatter D tiles: lane<16 -> M=r, N=lane; lane>=16 -> M=r+8, N=lane-16
      int nl0 = lane & 15;
      int mh  = (lane >> 4) << 3;
#pragma unroll
      for (int nt = 0; nt < 16; ++nt) {
        int nl = nt * 16 + nl0;
#pragma unroll
        for (int r = 0; r < 8; ++r)
          gates[(wave * 16 + r + mh) * 257 + nl] = acc[nt][r];
      }
    }
    __syncthreads();

    // ---- P3: elementwise LSTM cell + anomaly memory (each (b,j) owned by one fixed thread)
    const int   md = mode[t];
    const int   hd = headArr[t];
    const float wn = wnorm[t];
    float dotacc = 0.f;
#pragma unroll 4
    for (int k = 0; k < 32; ++k) {
      const int j   = jbase + (k << 3);
      const int idx = bloc * 257 + j;
      float ig = sigmoidf_(gates[(0 * 16 + bloc) * 257 + j] + biasS[j]);
      float fg = sigmoidf_(gates[(1 * 16 + bloc) * 257 + j] + biasS[256 + j]);
      float gg = tanhf   (gates[(2 * 16 + bloc) * 257 + j] + biasS[512 + j]);
      float og = sigmoidf_(gates[(3 * 16 + bloc) * 257 + j] + biasS[768 + j]);
      float c  = cbuf[idx];
      float nc = fg * c + ig * gg;
      float nh = og * tanhf(nc);
      float cn;
      if (md == 0) {                                  // anomalous: smooth toward memory blend
        float blend = Sbuf[idx] * 0.7f + nc * 0.3f;   // weighted*(1-DECAY) + new_c*DECAY
        cn = c * 0.9f + blend * 0.1f;                 // c*(1-SMOOTH) + blend*SMOOTH
      } else if (md == 2) {                           // full buffer: O(1) roll+append
        float oldest = ring[(hd * 16 + bloc) * 257 + j];
        Sbuf[idx] = (Sbuf[idx] - w0c * oldest) * inv_e + w9c * nc;
        ring[(hd * 16 + bloc) * 257 + j] = nc;
        cn = nc;
      } else {                                        // warmup append: recompute small sum
        ring[(hd * 16 + bloc) * 257 + j] = nc;
        float S = 0.f, ei = 1.f;
        for (int i = 0; i <= hd; ++i) { S += ei * ring[(i * 16 + bloc) * 257 + j]; ei *= E_F; }
        Sbuf[idx] = S * wn;
        cn = nc;
      }
      cbuf[idx] = cn;
      dotacc += nh * woutS[j];
      {  // write nh into hA (A-fragment bf16 layout) for next step's WMMA
        int kt = j >> 5, kk = j & 31;
        int lh = kk >> 4;
        int e  = kk & 15;
        int ln = bloc + (lh << 4);
        hA[((kt * 32 + ln) << 4) + e] = (__bf16)nh;
      }
      if (t == TT - 1) {
        hOut[(size_t)(wgb + bloc) * HH + j] = nh;
        cOut[(size_t)(wgb + bloc) * HH + j] = cn;
      }
    }
    atomicAdd(&outacc[bloc], dotacc);   // ds_add_f32 reduction for out_t
    __syncthreads();
    if (tid < 16) out[(size_t)(wgb + tid) * TT + t] = outacc[tid] + b0;
    __syncthreads();
  }
}

extern "C" void kernel_launch(void* const* d_in, const int* in_sizes, int n_in,
                              void* d_out, int out_size, void* d_ws, size_t ws_size,
                              hipStream_t stream) {
  const float* x    = (const float*)d_in[0];
  const float* Wih  = (const float*)d_in[1];
  const float* bih  = (const float*)d_in[2];
  const float* Whh  = (const float*)d_in[3];
  const float* bhh  = (const float*)d_in[4];
  const float* Wout = (const float*)d_in[5];
  const float* bout = (const float*)d_in[6];

  char* ws = (char*)d_ws;
  __bf16* Bhh  = (__bf16*)(ws + BHH_OFF);
  __bf16* Bih  = (__bf16*)(ws + BIH_OFF);
  int*    anom = (int*)  (ws + ANOM_OFF);
  int*    mode = (int*)  (ws + MODE_OFF);
  int*    head = (int*)  (ws + HEAD_OFF);
  float*  wnrm = (float*)(ws + WNORM_OFF);
  float*  bias = (float*)(ws + BIAS_OFF);

  float* out  = (float*)d_out;
  float* hOut = out + (size_t)BB * TT;
  float* cOut = hOut + (size_t)BB * HH;

  pack_whh <<<dim3(64, 8), 32, 0, stream>>>(Whh, Bhh);
  pack_wih <<<64, 32, 0, stream>>>(Wih, Bih);
  bias_sum <<<4, 256, 0, stream>>>(bih, bhh, bias);
  anom_scan<<<TT, BB, 0, stream>>>(x, anom);
  schedule_k<<<1, 1, 0, stream>>>(anom, mode, head, wnrm);

  // LDS: ring 164480 + gates 65792 + c 16448 + S 16448 + xs 1024 + wout 1024
  //      + bias 4096 + outacc 64 + hA 8192 = 277568 bytes (< 320 KB/WGP)
  size_t shmem = (size_t)(67344 * 4 + 8192);
  lstm_main<<<16, 128, shmem, stream>>>(x, Bhh, Bih, mode, head, wnrm, bias,
                                        Wout, bout, out, hOut, cOut);
}
